// GNNAK_69028714381398
// MI455X (gfx1250) — matrix operand order repaired and checked by reading, not compiled, not run on hardware
//
#include <hip/hip_runtime.h>
#include <hip/hip_bf16.h>

// ---------------------------------------------------------------------------
// CDNA5 (gfx1250) nested-GNN. All dense projections via v_wmma_f32_16x16x32_f16
// (f16 weights pre-converted once, f32 activations converted on load, f32 acc).
// Each wave computes a 16-row x Nc output strip: one A fragment feeds Nc/16
// consecutive WMMAs. GRU and pooling-gate epilogues fused into the GEMMs.
// ---------------------------------------------------------------------------

typedef __attribute__((ext_vector_type(16))) _Float16 v16h;
typedef __attribute__((ext_vector_type(8)))  float    v8f;

#define DD 64

__device__ __forceinline__ float sigm(float v) { return 1.0f / (1.0f + __expf(-v)); }

// A-matrix 16x32 f16 fragment: lane&15 = row M; lanes 0-15 K block 0-15,
// lanes 16-31 K block 16-31. Element i: K = k0 + hi16 + i.
// p points at A_row + k0 + hi16 (64B aligned: K%32==0, hi16 in {0,16}).
__device__ __forceinline__ v16h load_a_frag(const float* __restrict__ p) {
  const float4* q = (const float4*)p;
  float4 a0 = q[0], a1 = q[1], a2 = q[2], a3 = q[3];
  v16h v;
  v[0]  = (_Float16)a0.x; v[1]  = (_Float16)a0.y; v[2]  = (_Float16)a0.z; v[3]  = (_Float16)a0.w;
  v[4]  = (_Float16)a1.x; v[5]  = (_Float16)a1.y; v[6]  = (_Float16)a1.z; v[7]  = (_Float16)a1.w;
  v[8]  = (_Float16)a2.x; v[9]  = (_Float16)a2.y; v[10] = (_Float16)a2.z; v[11] = (_Float16)a2.w;
  v[12] = (_Float16)a3.x; v[13] = (_Float16)a3.y; v[14] = (_Float16)a3.z; v[15] = (_Float16)a3.w;
  return v;
}

// B-matrix 32x16 f16 fragment: lane&15 = column; lanes 0-15 K block 0-15,
// lanes 16-31 K block 16-31. W is f16 [Nc,K] row-major -> contiguous 32B.
__device__ __forceinline__ v16h load_b_frag(const _Float16* __restrict__ p) {
  return *(const v16h*)p;
}

__device__ __forceinline__ v16h wmma_step(v16h a, v16h b, v8f& acc) {
  acc = __builtin_amdgcn_wmma_f32_16x16x32_f16(false, a, false, b, (short)0,
                                               acc, false, false);
  return a;
}

// ----- Generic strip GEMM: C[M,NT*16] = act(A[M,K] @ W16^T + bias) ----------
// ACT: 0 none, 1 sigmoid, 2 relu, 3 elu. One 16 x (NT*16) strip per wave.
template <int NT, int ACT>
__global__ void gemm_wmma(const float* __restrict__ A, const _Float16* __restrict__ W,
                          const float* __restrict__ bias, float* __restrict__ C,
                          int M, int K) {
  const int lane = threadIdx.x & 31;
  const int mTile = blockIdx.x * 8 + (threadIdx.x >> 5);
  if (mTile >= (M >> 4)) return;                 // wave-uniform: EXEC stays full
  constexpr int Nc = NT * 16;
  const int half8 = (lane >> 4) << 3;
  const int hi16  = (lane >> 4) << 4;
  const int row   = (mTile << 4) + (lane & 15);
  const int cb    = lane & 15;
  const float* arow = A + (size_t)row * K;

  v8f acc[NT] = {};
  for (int k0 = 0; k0 < K; k0 += 32) {
    v16h a = load_a_frag(arow + k0 + hi16);
#pragma unroll
    for (int t = 0; t < NT; ++t) {
      v16h b = load_b_frag(W + (size_t)(t * 16 + cb) * K + k0 + hi16);
      wmma_step(a, b, acc[t]);
    }
  }
#pragma unroll
  for (int t = 0; t < NT; ++t) {
    const int col = t * 16 + cb;
    const float bv = bias ? bias[col] : 0.0f;
#pragma unroll
    for (int v = 0; v < 8; ++v) {
      int r = (mTile << 4) + v + ((lane >> 4) << 3);
      float val = acc[t][v] + bv;
      if (ACT == 1) val = sigm(val);
      if (ACT == 2) val = fmaxf(val, 0.0f);
      if (ACT == 3) val = val > 0.0f ? val : __expf(val) - 1.0f;
      C[(size_t)r * Nc + col] = val;
    }
  }
}

// ----- tbc: x = concat([x, zl]) @ W^T + b, written in place (K=128, Nc=64) --
// Each 32-wide K block lies wholly in x (k<64) or zl (k>=64): uniform select.
__global__ void gemm_tbc(float* __restrict__ x, const float* __restrict__ zl,
                         const _Float16* __restrict__ W, const float* __restrict__ bias,
                         int M) {
  const int lane = threadIdx.x & 31;
  const int mTile = blockIdx.x * 8 + (threadIdx.x >> 5);
  if (mTile >= (M >> 4)) return;
  const int half8 = (lane >> 4) << 3;
  const int hi16  = (lane >> 4) << 4;
  const int row   = (mTile << 4) + (lane & 15);
  const int cb    = lane & 15;
  const float* xrow = x  + (size_t)row * DD;
  const float* zrow = zl + (size_t)row * DD;

  v8f acc[4] = {};
#pragma unroll
  for (int k0 = 0; k0 < 128; k0 += 32) {
    const float* base = (k0 < 64) ? (xrow + k0) : (zrow + (k0 - 64));
    v16h a = load_a_frag(base + hi16);
#pragma unroll
    for (int t = 0; t < 4; ++t) {
      v16h b = load_b_frag(W + (size_t)(t * 16 + cb) * 128 + k0 + hi16);
      wmma_step(a, b, acc[t]);
    }
  }
#pragma unroll
  for (int t = 0; t < 4; ++t) {
    const int col = t * 16 + cb;
    const float bv = bias[col];
#pragma unroll
    for (int v = 0; v < 8; ++v) {
      int r = (mTile << 4) + v + ((lane >> 4) << 3);
      x[(size_t)r * DD + col] = acc[t][v] + bv;  // per-wave rows disjoint; safe
    }
  }
}

// ----- Fused GRU: gi = agg @ Wih^T (+bih) computed in-registers, then the
// full GRU cell using gh (precomputed, bias included) and x, writing x. ------
__global__ void gemm_gru(const float* __restrict__ agg, const _Float16* __restrict__ Wih,
                         const float* __restrict__ bih, const float* __restrict__ gh,
                         float* __restrict__ x, int M) {
  const int lane = threadIdx.x & 31;
  const int mTile = blockIdx.x * 8 + (threadIdx.x >> 5);
  if (mTile >= (M >> 4)) return;
  const int half8 = (lane >> 4) << 3;
  const int hi16  = (lane >> 4) << 4;
  const int row   = (mTile << 4) + (lane & 15);
  const int cb    = lane & 15;
  const float* arow = agg + (size_t)row * DD;

  v8f acc[12] = {};
#pragma unroll
  for (int k0 = 0; k0 < DD; k0 += 32) {
    v16h a = load_a_frag(arow + k0 + hi16);
#pragma unroll
    for (int t = 0; t < 12; ++t) {
      v16h b = load_b_frag(Wih + (size_t)(t * 16 + cb) * DD + k0 + hi16);
      wmma_step(a, b, acc[t]);
    }
  }
  // r/u/c parts for feature d = t*16+cb live in acc[t], acc[t+4], acc[t+8]
#pragma unroll
  for (int t = 0; t < 4; ++t) {
    const int d = t * 16 + cb;
    const float br = bih[d], bu = bih[DD + d], bc = bih[2 * DD + d];
#pragma unroll
    for (int v = 0; v < 8; ++v) {
      int r = (mTile << 4) + v + ((lane >> 4) << 3);
      size_t gb = (size_t)r * 3 * DD;
      float rr = sigm(acc[t][v]     + br + gh[gb + d]);
      float uu = sigm(acc[t + 4][v] + bu + gh[gb + DD + d]);
      float cc = tanhf(acc[t + 8][v] + bc + rr * gh[gb + 2 * DD + d]);
      size_t xi = (size_t)r * DD + d;
      x[xi] = (1.0f - uu) * cc + uu * x[xi];
    }
  }
}

// ----- Fused pooling gates: gs = sig(zl@Wsg^T+b), gc = sig(zl@Wcg^T+b), then
// atomic scatter of gs*x -> hcat[:,0:64] (subgraph) and gc*x -> hcat[:,128:192]
// (original-node), never materializing gs/gc. ------------------------------
__global__ void gemm_gates(const float* __restrict__ zl, const _Float16* __restrict__ Wsg,
                           const _Float16* __restrict__ Wcg, const float* __restrict__ sgb,
                           const float* __restrict__ cgb, const float* __restrict__ x,
                           const int* __restrict__ sub, const int* __restrict__ orig,
                           float* __restrict__ hcat, int M) {
  const int lane = threadIdx.x & 31;
  const int mTile = blockIdx.x * 8 + (threadIdx.x >> 5);
  if (mTile >= (M >> 4)) return;
  const int half8 = (lane >> 4) << 3;
  const int hi16  = (lane >> 4) << 4;
  const int row   = (mTile << 4) + (lane & 15);
  const int cb    = lane & 15;
  const float* arow = zl + (size_t)row * DD;

  v8f acc[8] = {};
#pragma unroll
  for (int k0 = 0; k0 < DD; k0 += 32) {
    v16h a = load_a_frag(arow + k0 + hi16);
#pragma unroll
    for (int t = 0; t < 8; ++t) {
      const _Float16* Wt = (t < 4) ? (Wsg + (size_t)(t * 16 + cb) * DD)
                                   : (Wcg + (size_t)((t - 4) * 16 + cb) * DD);
      v16h b = load_b_frag(Wt + k0 + hi16);
      wmma_step(a, b, acc[t]);
    }
  }
#pragma unroll
  for (int t = 0; t < 4; ++t) {
    const int d = t * 16 + cb;
    const float bs = sgb[d], bc2 = cgb[d];
#pragma unroll
    for (int v = 0; v < 8; ++v) {
      int r = (mTile << 4) + v + ((lane >> 4) << 3);
      float xv = x[(size_t)r * DD + d];
      float gsv = sigm(acc[t][v] + bs);
      float gcv = sigm(acc[t + 4][v] + bc2);
      atomicAdd(&hcat[(size_t)sub[r]  * 192 + d],       gsv * xv);
      atomicAdd(&hcat[(size_t)orig[r] * 192 + 128 + d], gcv * xv);
    }
  }
}

// ----- Weight conversion (once per launch; weights stay f16 + L2-hot) -------
__global__ void w_cvt(_Float16* __restrict__ dst, const float* __restrict__ src, int n) {
  int i = blockIdx.x * blockDim.x + threadIdx.x;
  if (i < n) dst[i] = (_Float16)src[i];
}
// conv_w is [L][K=64][Nc=64]; store transposed per layer -> [L][Nc][K]
__global__ void w_cvt_t64(_Float16* __restrict__ dst, const float* __restrict__ src, int n) {
  int idx = blockIdx.x * blockDim.x + threadIdx.x;
  if (idx >= n) return;
  int l = idx >> 12, r = idx & 4095, nn = r >> 6, k = r & 63;
  dst[idx] = (_Float16)src[(l << 12) + (k << 6) + nn];
}

// ----- Irregular kernels (float4 where possible) ----------------------------
__global__ void embed4(float4* __restrict__ zl, const float* __restrict__ emb,
                       const int* __restrict__ z, int n) {
  int idx = blockIdx.x * blockDim.x + threadIdx.x;
  if (idx >= n * 16) return;
  int i = idx >> 4, q = idx & 15;
  zl[idx] = ((const float4*)(emb + (size_t)z[i] * DD))[q];
}

__global__ void edge_scatter4(float* __restrict__ agg, const float* __restrict__ m,
                              const int* __restrict__ src, const int* __restrict__ dst,
                              int E) {
  int idx = blockIdx.x * blockDim.x + threadIdx.x;
  if (idx >= E * 16) return;
  int e = idx >> 4, q = idx & 15;
  float4 mv = ((const float4*)(m + (size_t)src[e] * DD))[q];
  float* ap = agg + (size_t)dst[e] * DD + q * 4;
  atomicAdd(ap + 0, mv.x); atomicAdd(ap + 1, mv.y);
  atomicAdd(ap + 2, mv.z); atomicAdd(ap + 3, mv.w);
}

__global__ void centers_kernel(int* __restrict__ centers, const int* __restrict__ sub,
                               int n) {
  int i = blockIdx.x * blockDim.x + threadIdx.x;
  if (i >= n) return;
  if (i == 0 || sub[i] != sub[i - 1]) centers[sub[i]] = i;
}

__global__ void xc_gather4(float* __restrict__ hcat, const float* __restrict__ x,
                           const int* __restrict__ centers, int S) {
  int idx = blockIdx.x * blockDim.x + threadIdx.x;
  if (idx >= S * 16) return;
  int s = idx >> 4, q = idx & 15;
  ((float4*)(hcat + (size_t)s * 192 + DD))[q] =
      ((const float4*)(x + (size_t)centers[s] * DD))[q];
}

__global__ void scatter_back4(float4* __restrict__ x, const float* __restrict__ h2,
                              const int* __restrict__ orig, int n) {
  int idx = blockIdx.x * blockDim.x + threadIdx.x;
  if (idx >= n * 16) return;
  int i = idx >> 4, q = idx & 15;
  x[idx] = ((const float4*)(h2 + (size_t)orig[i] * DD))[q];
}

__global__ void graph_sum(float* __restrict__ gout, const float* __restrict__ h2,
                          const int* __restrict__ s2g, int S) {
  int idx = blockIdx.x * blockDim.x + threadIdx.x;
  if (idx >= S * DD) return;
  int s = idx >> 6, d = idx & 63;
  atomicAdd(&gout[(size_t)s2g[s] * DD + d], h2[idx]);
}

__global__ void fc3_kernel(const float* __restrict__ f2, const float* __restrict__ w,
                           const float* __restrict__ b, float* __restrict__ out, int G) {
  int g = blockIdx.x * blockDim.x + threadIdx.x;
  if (g >= G) return;
  float s = b[0];
#pragma unroll
  for (int j = 0; j < 16; ++j) s += f2[g * 16 + j] * w[j];
  out[g] = s;
}

// ---------------------------------------------------------------------------
static inline int eb(long long cnt) { return (int)((cnt + 255) / 256); }

extern "C" void kernel_launch(void* const* d_in, const int* in_sizes, int n_in,
                              void* d_out, int out_size, void* d_ws, size_t ws_size,
                              hipStream_t stream) {
  const int N = in_sizes[0];
  const int E = in_sizes[1] / 2;
  const int S = in_sizes[4];
  const int G = out_size;          // reference output is [G, 1]

  const int* z    = (const int*)d_in[0];
  const int* src  = (const int*)d_in[1];
  const int* dst  = src + E;
  const int* sub  = (const int*)d_in[2];
  const int* orig = (const int*)d_in[3];
  const int* s2g  = (const int*)d_in[4];
  const float* z_emb   = (const float*)d_in[7];
  const float* tbc_w   = (const float*)d_in[8];
  const float* tbc_b   = (const float*)d_in[9];
  const float* conv_w  = (const float*)d_in[10];
  const float* gru_wih = (const float*)d_in[11];
  const float* gru_whh = (const float*)d_in[12];
  const float* gru_bih = (const float*)d_in[13];
  const float* gru_bhh = (const float*)d_in[14];
  const float* sg_w = (const float*)d_in[15];
  const float* sg_b = (const float*)d_in[16];
  const float* cg_w = (const float*)d_in[17];
  const float* cg_b = (const float*)d_in[18];
  const float* pw1 = (const float*)d_in[19];
  const float* pb1 = (const float*)d_in[20];
  const float* pw2 = (const float*)d_in[21];
  const float* pb2 = (const float*)d_in[22];
  const float* fc1w = (const float*)d_in[23];
  const float* fc1b = (const float*)d_in[24];
  const float* fc2w = (const float*)d_in[25];
  const float* fc2b = (const float*)d_in[26];
  const float* fc3w = (const float*)d_in[27];
  const float* fc3b = (const float*)d_in[28];
  float* out = (float*)d_out;

  // ---- workspace carve (256B aligned) ----
  char* wp = (char*)d_ws;
  auto carve = [&](size_t bytes) -> void* {
    void* p = (void*)wp;
    wp += (bytes + 255) & ~(size_t)255;
    return p;
  };
  const size_t nd = (size_t)N * DD * sizeof(float);
  float* x    = (float*)carve(nd);
  float* zl   = (float*)carve(nd);
  float* m    = (float*)carve(nd);
  float* agg  = (float*)carve(nd);
  float* gh   = (float*)carve((size_t)N * 3 * DD * sizeof(float));
  float* hcat = (float*)carve((size_t)S * 192 * sizeof(float));
  float* h1   = (float*)carve((size_t)S * DD * sizeof(float));
  float* h2   = (float*)carve((size_t)S * DD * sizeof(float));
  int*   centers = (int*)carve((size_t)S * sizeof(int));
  float* gout = (float*)carve((size_t)G * DD * sizeof(float));
  float* f1   = (float*)carve((size_t)G * 32 * sizeof(float));
  float* f2   = (float*)carve((size_t)G * 16 * sizeof(float));
  // f16 weights
  _Float16* tbc16  = (_Float16*)carve(5 * DD * 2 * DD * sizeof(_Float16));
  _Float16* conv16 = (_Float16*)carve(5 * DD * DD * sizeof(_Float16));
  _Float16* wih16  = (_Float16*)carve(5 * 3 * DD * DD * sizeof(_Float16));
  _Float16* whh16  = (_Float16*)carve(5 * 3 * DD * DD * sizeof(_Float16));
  _Float16* sg16   = (_Float16*)carve(6 * DD * DD * sizeof(_Float16));
  _Float16* cg16   = (_Float16*)carve(6 * DD * DD * sizeof(_Float16));
  _Float16* pw116  = (_Float16*)carve(6 * DD * 3 * DD * sizeof(_Float16));
  _Float16* pw216  = (_Float16*)carve(6 * DD * DD * sizeof(_Float16));
  _Float16* fc116  = (_Float16*)carve(32 * DD * sizeof(_Float16));
  _Float16* fc216  = (_Float16*)carve(16 * 32 * sizeof(_Float16));
  (void)ws_size; (void)n_in;

  auto cvt = [&](_Float16* d16, const float* s32, int cnt) {
    w_cvt<<<eb(cnt), 256, 0, stream>>>(d16, s32, cnt);
  };
  cvt(tbc16, tbc_w, 5 * DD * 2 * DD);
  w_cvt_t64<<<eb(5 * DD * DD), 256, 0, stream>>>(conv16, conv_w, 5 * DD * DD);
  cvt(wih16, gru_wih, 5 * 3 * DD * DD);
  cvt(whh16, gru_whh, 5 * 3 * DD * DD);
  cvt(sg16, sg_w, 6 * DD * DD);
  cvt(cg16, cg_w, 6 * DD * DD);
  cvt(pw116, pw1, 6 * DD * 3 * DD);
  cvt(pw216, pw2, 6 * DD * DD);
  cvt(fc116, fc1w, 32 * DD);
  cvt(fc216, fc2w, 16 * 32);

  auto gblocks = [](int M) { return ((M >> 4) + 7) / 8; };

  centers_kernel<<<eb(N), 256, 0, stream>>>(centers, sub, N);

  for (int l = 0; l < 5; ++l) {
    embed4<<<eb((long long)N * 16), 256, 0, stream>>>(
        (float4*)zl, z_emb + (size_t)l * 100 * DD, z, N);

    if (l == 0) {
      hipMemcpyAsync(x, zl, nd, hipMemcpyDeviceToDevice, stream);
    } else {
      gemm_tbc<<<gblocks(N), 256, 0, stream>>>(
          x, zl, tbc16 + (size_t)l * DD * 2 * DD, tbc_b + (size_t)l * DD, N);
    }

    // message passing: m = x @ conv_w[l]; agg = segment_sum(m[src], dst)
    gemm_wmma<4, 0><<<gblocks(N), 256, 0, stream>>>(
        x, conv16 + (size_t)l * DD * DD, nullptr, m, N, DD);
    hipMemsetAsync(agg, 0, nd, stream);
    edge_scatter4<<<eb((long long)E * 16), 256, 0, stream>>>(agg, m, src, dst, E);

    // GRU: gh materialized (bias folded); gi computed in-registers + fused cell
    gemm_wmma<12, 0><<<gblocks(N), 256, 0, stream>>>(
        x, whh16 + (size_t)l * 3 * DD * DD, gru_bhh + (size_t)l * 3 * DD, gh, N, DD);
    gemm_gru<<<gblocks(N), 256, 0, stream>>>(
        agg, wih16 + (size_t)l * 3 * DD * DD, gru_bih + (size_t)l * 3 * DD, gh, x, N);

    // pooling: fused gates + gated atomic scatter, center gather, pool MLP
    hipMemsetAsync(hcat, 0, (size_t)S * 192 * sizeof(float), stream);
    gemm_gates<<<gblocks(N), 256, 0, stream>>>(
        zl, sg16 + (size_t)l * DD * DD, cg16 + (size_t)l * DD * DD,
        sg_b + (size_t)l * DD, cg_b + (size_t)l * DD, x, sub, orig, hcat, N);
    xc_gather4<<<eb((long long)S * 16), 256, 0, stream>>>(hcat, x, centers, S);
    gemm_wmma<4, 2><<<gblocks(S), 256, 0, stream>>>(
        hcat, pw116 + (size_t)l * DD * 3 * DD, pb1 + (size_t)l * DD, h1, S, 3 * DD);
    gemm_wmma<4, 0><<<gblocks(S), 256, 0, stream>>>(
        h1, pw216 + (size_t)l * DD * DD, pb2 + (size_t)l * DD, h2, S, DD);
    scatter_back4<<<eb((long long)N * 16), 256, 0, stream>>>((float4*)x, h2, orig, N);
  }

  // final pooling (l = 5) reduced to graphs
  {
    const int l = 5;
    embed4<<<eb((long long)N * 16), 256, 0, stream>>>(
        (float4*)zl, z_emb + (size_t)l * 100 * DD, z, N);
    hipMemsetAsync(hcat, 0, (size_t)S * 192 * sizeof(float), stream);
    gemm_gates<<<gblocks(N), 256, 0, stream>>>(
        zl, sg16 + (size_t)l * DD * DD, cg16 + (size_t)l * DD * DD,
        sg_b + (size_t)l * DD, cg_b + (size_t)l * DD, x, sub, orig, hcat, N);
    xc_gather4<<<eb((long long)S * 16), 256, 0, stream>>>(hcat, x, centers, S);
    gemm_wmma<4, 2><<<gblocks(S), 256, 0, stream>>>(
        hcat, pw116 + (size_t)l * DD * 3 * DD, pb1 + (size_t)l * DD, h1, S, 3 * DD);
    gemm_wmma<4, 0><<<gblocks(S), 256, 0, stream>>>(
        h1, pw216 + (size_t)l * DD * DD, pb2 + (size_t)l * DD, h2, S, DD);
    hipMemsetAsync(gout, 0, (size_t)G * DD * sizeof(float), stream);
    graph_sum<<<eb((long long)S * DD), 256, 0, stream>>>(gout, h2, s2g, S);
  }

  // head: elu(fc1) -> elu(fc2) -> fc3
  gemm_wmma<2, 3><<<gblocks(G), 256, 0, stream>>>(gout, fc116, fc1b, f1, G, DD);
  gemm_wmma<1, 3><<<gblocks(G), 256, 0, stream>>>(f1, fc216, fc2b, f2, G, 32);
  fc3_kernel<<<eb(G), 64, 0, stream>>>(f2, fc3w, fc3b, out, G);
}
